// MobileViTv2LinearSelfAttention_32581621907579
// MI455X (gfx1250) — compile-verified, matrix operand order checked by reading, not run on hardware
//
#include <hip/hip_runtime.h>
#include <hip/hip_bf16.h>
#include <stdint.h>

#define C_DIM 512
#define P_DIM 4
#define N_DIM 1024
#define ROW_STRIDE  (P_DIM * N_DIM)          /* stride between channels, floats */
#define SLAB_STRIDE (C_DIM * P_DIM * N_DIM)  /* stride between batches, floats */

typedef __attribute__((ext_vector_type(16))) __bf16 bf16x16;
typedef __attribute__((ext_vector_type(8)))  float  f32x8;

static __device__ __forceinline__ uint16_t f2bf(float f) {
  union { float f; uint32_t u; } c; c.f = f;
  uint32_t r = c.u + 0x7FFFu + ((c.u >> 16) & 1u);   // round-to-nearest-even
  return (uint16_t)(r >> 16);
}
static __device__ __forceinline__ uint32_t pack2(uint16_t lo, uint16_t hi) {
  return (uint32_t)lo | ((uint32_t)hi << 16);
}

// ---------------------------------------------------------------------------
// Kernel 0: convert W_v / W_out to bf16 and swizzle into the WMMA A-fragment
// layout (ISA 7.12.2, 16-bit A 16x32):
//   element (m, k): kc = k>>5, kk = k&31
//   half = (kk>>3)&1, j = ((kk>>4)&1)*4 + ((kk&7)>>1), lohi = kk&1
//   dst[(kc*512 + m)*32 + half*16 + j*2 + lohi]
// A lane (half,lm) then reads its 8 fragment dwords as two contiguous b128
// loads; a wave covers a contiguous 1 KB region.
// ---------------------------------------------------------------------------
__global__ void __launch_bounds__(256) prep_weights(
    const float* __restrict__ w_qkv, const float* __restrict__ w_out,
    uint16_t* __restrict__ wv_pre, uint16_t* __restrict__ wout_pre) {
  const int t = threadIdx.x;
  const int m = blockIdx.x;              // 0..511
  const int sel = blockIdx.y;            // 0 = W_v, 1 = W_out
  const float* src = sel ? (w_out + (size_t)m * C_DIM)
                         : (w_qkv + (size_t)(1 + C_DIM + m) * C_DIM);
  uint16_t* dst = sel ? wout_pre : wv_pre;
#pragma unroll
  for (int i = 0; i < 2; ++i) {
    const int k = t + i * 256;
    const int kc = k >> 5, kk = k & 31;
    const int half = (kk >> 3) & 1;
    const int j = ((kk >> 4) & 1) * 4 + ((kk & 7) >> 1);
    dst[(size_t)(kc * C_DIM + m) * 32 + half * 16 + j * 2 + (kk & 1)] = f2bf(src[k]);
  }
}

// ---------------------------------------------------------------------------
// Kernel 1: per (b,p) slab compute scores + context vector.
//   ctx = W_k @ (X @ softmax(w_q·X + b_q)) + b_k      (uses sum(scores)==1)
// ---------------------------------------------------------------------------
__global__ void __launch_bounds__(256) ctx_kernel(
    const float* __restrict__ x, const float* __restrict__ w_qkv,
    const float* __restrict__ b_qkv, float* __restrict__ ctx_ws) {
  __shared__ float sc[N_DIM];
  __shared__ float xs[C_DIM];
  __shared__ float red[256];
  const int t = threadIdx.x;
  const int slab = blockIdx.x;
  const int b = slab >> 2, p = slab & 3;
  const size_t xbase = (size_t)b * SLAB_STRIDE + (size_t)p * N_DIM;

  float a0 = 0.f, a1 = 0.f, a2 = 0.f, a3 = 0.f;
  for (int c = 0; c < C_DIM; ++c) {
    const float w = w_qkv[c];
    const float* xr = x + xbase + (size_t)c * ROW_STRIDE;
    a0 += w * xr[t];       a1 += w * xr[t + 256];
    a2 += w * xr[t + 512]; a3 += w * xr[t + 768];
  }
  const float bq = b_qkv[0];
  sc[t] = a0 + bq; sc[t + 256] = a1 + bq; sc[t + 512] = a2 + bq; sc[t + 768] = a3 + bq;
  __syncthreads();

  red[t] = fmaxf(fmaxf(sc[t], sc[t + 256]), fmaxf(sc[t + 512], sc[t + 768]));
  __syncthreads();
  for (int s = 128; s > 0; s >>= 1) { if (t < s) red[t] = fmaxf(red[t], red[t + s]); __syncthreads(); }
  const float mx = red[0];
  __syncthreads();
  float e0 = __expf(sc[t] - mx),       e1 = __expf(sc[t + 256] - mx);
  float e2 = __expf(sc[t + 512] - mx), e3 = __expf(sc[t + 768] - mx);
  red[t] = e0 + e1 + e2 + e3;
  __syncthreads();
  for (int s = 128; s > 0; s >>= 1) { if (t < s) red[t] += red[t + s]; __syncthreads(); }
  const float inv = 1.0f / red[0];
  __syncthreads();
  sc[t] = e0 * inv; sc[t + 256] = e1 * inv; sc[t + 512] = e2 * inv; sc[t + 768] = e3 * inv;
  __syncthreads();

  for (int c = t; c < C_DIM; c += 256) {
    const float* xr = x + xbase + (size_t)c * ROW_STRIDE;
    float a = 0.f;
    for (int n = 0; n < N_DIM; ++n) a += xr[n] * sc[n];
    xs[c] = a;
  }
  __syncthreads();

  for (int o = t; o < C_DIM; o += 256) {
    const float* wr = w_qkv + (size_t)(1 + o) * C_DIM;
    float a = 0.f;
    for (int c = 0; c < C_DIM; ++c) a += wr[c] * xs[c];
    ctx_ws[(size_t)slab * C_DIM + o] = a + b_qkv[1 + o];
  }
}

// ---------------------------------------------------------------------------
// Kernel 2: fused  out = W_out @ (relu(W_v @ X + b_v) * ctx) + b_out
// One block = one (slab, 64-column tile); 16 waves, each owns a 32x64 tile
// (2x4 WMMA accumulators). A fragments stream straight from pre-swizzled
// global bf16 weights (2 x global_load_b128 per fragment, wave-contiguous).
// B operands live in LDS in fragment layout (2 x ds_load_b128 per fragment).
// Xs is double-buffered: one barrier per K-chunk, next chunk's X float4 load
// issued before the barrier so HBM fetch overlaps the WMMA burst.
// LDS (bytes): Mbuf 64x520u16 @0 | Xs 2x(64x40u16) @66560 | ctx/bv/bout @76800
// ---------------------------------------------------------------------------
#define SMEM2 (66560 + 10240 + 3 * 512 * 4)

__global__ void __launch_bounds__(512) fused_kernel(
    const float* __restrict__ x, const float* __restrict__ b_qkv,
    const float* __restrict__ b_out,
    const uint16_t* __restrict__ wv_pre, const uint16_t* __restrict__ wout_pre,
    const float* __restrict__ ctx_ws, float* __restrict__ out) {
  extern __shared__ char smem[];
  uint16_t* Mbuf   = (uint16_t*)smem;            // [col] stride 520: [kc][half][j][lohi]
  uint32_t* Mbuf_u = (uint32_t*)smem;
  uint16_t* Xs     = (uint16_t*)(smem + 66560);  // 2 buffers, [col] stride 40
  float* ctxs  = (float*)(smem + 66560 + 10240);
  float* bvs   = ctxs + 512;
  float* bouts = bvs + 512;

  const int t = threadIdx.x;
  const int lane = t & 31, wv = t >> 5;
  const int half = lane >> 4, lm = lane & 15;
  const int slab = blockIdx.y;
  const int b = slab >> 2, p = slab & 3;
  const int n0 = blockIdx.x * 64;
  const size_t xbase = (size_t)b * SLAB_STRIDE + (size_t)p * N_DIM;

  ctxs[t]  = ctx_ws[(size_t)slab * C_DIM + t];
  bvs[t]   = b_qkv[1 + C_DIM + t];
  bouts[t] = b_out[t];

  f32x8 acc[2][4];
#pragma unroll
  for (int mt = 0; mt < 2; ++mt)
#pragma unroll
    for (int nt = 0; nt < 4; ++nt) acc[mt][nt] = (f32x8){0.f,0.f,0.f,0.f,0.f,0.f,0.f,0.f};

  union Frag { uint32_t u[8]; uint4 q[2]; bf16x16 v; };

  // per-thread X staging geometry (fragment-layout store position)
  const int kk  = t >> 4;            // 0..31
  const int c0  = (t & 15) * 4;      // 0..60
  const int xoff = (((kk >> 3) & 1) * 16) + ((((kk >> 4) & 1) * 4 + ((kk & 7) >> 1)) * 2) + (kk & 1);
  const float* xcol = x + xbase + (size_t)kk * ROW_STRIDE + n0 + c0;

  // -------- Phase 1: V-tile = W_v @ X_tile  (bf16 WMMA, f32 accum) ---------
  float4 xv = *(const float4*)xcol;                 // chunk 0 prefetch
  for (int kc = 0; kc < 16; ++kc) {
    uint16_t* Xb = Xs + (kc & 1) * 2560;
    Xb[(c0 + 0) * 40 + xoff] = f2bf(xv.x);
    Xb[(c0 + 1) * 40 + xoff] = f2bf(xv.y);
    Xb[(c0 + 2) * 40 + xoff] = f2bf(xv.z);
    Xb[(c0 + 3) * 40 + xoff] = f2bf(xv.w);
    if (kc < 15)                                    // next chunk in flight during WMMAs
      xv = *(const float4*)(xcol + (size_t)(kc + 1) * 32 * ROW_STRIDE);
    Frag af[2];
#pragma unroll
    for (int mt = 0; mt < 2; ++mt) {                // A frags: no LDS dep, overlap barrier
      const int m = wv * 32 + mt * 16 + lm;
      const uint4* ap = (const uint4*)(wv_pre + (size_t)(kc * C_DIM + m) * 32 + half * 16);
      af[mt].q[0] = ap[0];
      af[mt].q[1] = ap[1];
    }
    __syncthreads();
    Frag bb[4];
#pragma unroll
    for (int nt = 0; nt < 4; ++nt) {
      const uint4* bp = (const uint4*)(Xb + (nt * 16 + lm) * 40 + half * 16);
      bb[nt].q[0] = bp[0];
      bb[nt].q[1] = bp[1];
    }
#pragma unroll
    for (int nt = 0; nt < 4; ++nt)
#pragma unroll
      for (int mt = 0; mt < 2; ++mt)
        acc[mt][nt] = __builtin_amdgcn_wmma_f32_16x16x32_bf16(
            false, af[mt].v, false, bb[nt].v, (short)0, acc[mt][nt], false, false);
  }
  __syncthreads();

  // epilogue 1: + b_v, relu, * ctx  -> Mbuf in fragment layout
  // value (m = wv*32 + mt*16 + half*8 + r, col): kc2=wv, half2=half,
  // j = mt*4 + (r>>1)  -> uint idx = col*260 + wv*16 + half*8 + mt*4 + (r>>1)
#pragma unroll
  for (int mt = 0; mt < 2; ++mt) {
    const int mbase = wv * 32 + mt * 16 + half * 8;
#pragma unroll
    for (int nt = 0; nt < 4; ++nt) {
      const int col = nt * 16 + lm;
#pragma unroll
      for (int r = 0; r < 8; r += 2) {
        float v0 = fmaxf(acc[mt][nt][r]     + bvs[mbase + r],     0.f) * ctxs[mbase + r];
        float v1 = fmaxf(acc[mt][nt][r + 1] + bvs[mbase + r + 1], 0.f) * ctxs[mbase + r + 1];
        Mbuf_u[col * 260 + wv * 16 + half * 8 + mt * 4 + (r >> 1)] = pack2(f2bf(v0), f2bf(v1));
      }
    }
  }
#pragma unroll
  for (int mt = 0; mt < 2; ++mt)
#pragma unroll
    for (int nt = 0; nt < 4; ++nt) acc[mt][nt] = (f32x8){0.f,0.f,0.f,0.f,0.f,0.f,0.f,0.f};
  __syncthreads();

  // -------- Phase 2: out-tile = W_out @ Mbuf (no staging, no barriers) -----
  for (int kc = 0; kc < 16; ++kc) {
    Frag af[2], bb[4];
#pragma unroll
    for (int mt = 0; mt < 2; ++mt) {
      const int m = wv * 32 + mt * 16 + lm;
      const uint4* ap = (const uint4*)(wout_pre + (size_t)(kc * C_DIM + m) * 32 + half * 16);
      af[mt].q[0] = ap[0];
      af[mt].q[1] = ap[1];
    }
#pragma unroll
    for (int nt = 0; nt < 4; ++nt) {
      const uint4* bp = (const uint4*)(Mbuf + (nt * 16 + lm) * 520 + kc * 32 + half * 16);
      bb[nt].q[0] = bp[0];
      bb[nt].q[1] = bp[1];
    }
#pragma unroll
    for (int nt = 0; nt < 4; ++nt)
#pragma unroll
      for (int mt = 0; mt < 2; ++mt)
        acc[mt][nt] = __builtin_amdgcn_wmma_f32_16x16x32_bf16(
            false, af[mt].v, false, bb[nt].v, (short)0, acc[mt][nt], false, false);
  }

  // epilogue 2: + b_out, fp32 store to HBM (lanes cover consecutive n)
#pragma unroll
  for (int mt = 0; mt < 2; ++mt) {
    const int mbase = wv * 32 + mt * 16 + half * 8;
#pragma unroll
    for (int nt = 0; nt < 4; ++nt) {
      const int n = n0 + nt * 16 + lm;
#pragma unroll
      for (int r = 0; r < 8; ++r) {
        const int m = mbase + r;
        out[xbase + (size_t)m * ROW_STRIDE + n] = acc[mt][nt][r] + bouts[m];
      }
    }
  }
}

// ---------------------------------------------------------------------------
extern "C" void kernel_launch(void* const* d_in, const int* in_sizes, int n_in,
                              void* d_out, int out_size, void* d_ws, size_t ws_size,
                              hipStream_t stream) {
  (void)in_sizes; (void)n_in; (void)out_size; (void)ws_size;
  const float* x     = (const float*)d_in[0];
  const float* w_qkv = (const float*)d_in[1];
  const float* b_qkv = (const float*)d_in[2];
  const float* w_out = (const float*)d_in[3];
  const float* b_out = (const float*)d_in[4];
  float* out = (float*)d_out;

  // workspace carve: ctx (64*512 f32 = 128 KB) | wv_pre (512 KB) | wout_pre (512 KB)
  float*    ctx_ws   = (float*)d_ws;
  uint16_t* wv_pre   = (uint16_t*)((char*)d_ws + 64 * C_DIM * sizeof(float));
  uint16_t* wout_pre = wv_pre + (size_t)C_DIM * C_DIM;

  prep_weights<<<dim3(512, 2), dim3(256), 0, stream>>>(w_qkv, w_out, wv_pre, wout_pre);
  ctx_kernel<<<dim3(64), dim3(256), 0, stream>>>(x, w_qkv, b_qkv, ctx_ws);

  (void)hipFuncSetAttribute(reinterpret_cast<const void*>(fused_kernel),
                            hipFuncAttributeMaxDynamicSharedMemorySize, SMEM2);
  fused_kernel<<<dim3(16, 64), dim3(512), SMEM2, stream>>>(
      x, b_qkv, b_out, wv_pre, wout_pre, ctx_ws, out);
}
